// Linformer_62225486184477
// MI455X (gfx1250) — matrix operand order repaired and checked by reading, not compiled, not run on hardware
//
#include <hip/hip_runtime.h>
#include <hip/hip_bf16.h>

#define DEV static __device__ __forceinline__

typedef __attribute__((ext_vector_type(16))) __bf16          v16bf;
typedef __attribute__((ext_vector_type(16))) unsigned short  v16u;
typedef __attribute__((ext_vector_type(8)))  unsigned short  v8u;
typedef __attribute__((ext_vector_type(8)))  float           v8f;
typedef __attribute__((ext_vector_type(4)))  unsigned int    v4ui;
typedef __attribute__((ext_vector_type(8)))  int             v8i;
typedef __attribute__((ext_vector_type(4)))  int             v4i;

// Problem constants (fixed by the reference).
static constexpr int Bb = 4, Nn = 4096, Cc = 1024, Hh = 16, Dd = 64, KD = 256, FFd = 4096, LL = 2;

DEV unsigned short f2bf(float f){
  unsigned int u = __builtin_bit_cast(unsigned int, f);
  u += 0x7fffu + ((u >> 16) & 1u);           // round-to-nearest-even
  return (unsigned short)(u >> 16);
}

DEV v8f v8f_zero(){ v8f z = {0.f,0.f,0.f,0.f,0.f,0.f,0.f,0.f}; return z; }

DEV v8f wmma_bf16(v16u a, v16u b, v8f c){
  return __builtin_amdgcn_wmma_f32_16x16x32_bf16(
      false, __builtin_bit_cast(v16bf, a),
      false, __builtin_bit_cast(v16bf, b),
      (short)0, c, false, false);
}

// A fragment (16x32, row-major, k contiguous). Layout:
// lane holds row m, halves j -> k = 16*hi + j (contiguous).
DEV v16u load_a_frag(const unsigned short* A, long lda, int m0, int k0, int lid){
  int lm = lid & 15, hi = lid >> 4;
  const unsigned short* p = A + (long)(m0 + lm) * lda + k0 + 16 * hi;
  v8u lo  = *(const v8u*)p;
  v8u hiw = *(const v8u*)(p + 8);
  v16u r;
#pragma unroll
  for (int i = 0; i < 8; ++i){ r[i] = lo[i]; r[8 + i] = hiw[i]; }
  return r;
}

// B fragment "T flavor": W is (ncols x K) row-major; B(k,n) = W[n][k].
// Lane holds col n, halves j -> k = 16*hi + j (contiguous 32B).
DEV v16u load_bT_frag(const unsigned short* W, long ldw, int n0, int k0, int lid){
  int ln = lid & 15, hi = lid >> 4;
  return *(const v16u*)(W + (long)(n0 + ln) * ldw + k0 + 16 * hi);
}

// TDM: DMA a 2-D bf16 tile (tile_d1 rows x tile_d0 cols, row stride in elems)
// from global into LDS, with LDS row padding of 16B per 128B row
// (pad_interval=4 -> after 16*8B stored, pad_amount=3 -> 4 DWORDs), i.e.
// an LDS pitch of 144B = 72 halves. D# packed per CDNA5 ISA section 8.
DEV void tdm_load_tile(unsigned lds_addr, unsigned long long gaddr,
                       unsigned tensor_d0, unsigned tensor_d1,
                       unsigned tile_d0, unsigned tile_d1,
                       unsigned long long row_stride_elems)
{
  v4ui g0;
  g0[0] = 1u;                                            // count=1 (valid), user mode
  g0[1] = lds_addr;                                      // [63:32] lds_addr (bytes)
  g0[2] = (unsigned)(gaddr & 0xffffffffull);             // [95:64] global_addr lo
  g0[3] = (unsigned)((gaddr >> 32) & 0x1ffffffull)       // [120:96] global_addr hi
        | (2u << 30);                                    // [127:126] type=2 ("image")

  unsigned d0 = (1u << 16)                               // data_size = 2 bytes
              | (1u << 20)                               // pad_enable
              | (4u << 22)                               // pad_interval: 16*8B = 128B
              | (3u << 25);                              // pad_amount: 4 DWORDs = 16B
  v8i g1;
  g1[0] = (int)d0;
  g1[1] = (int)((tensor_d0 & 0xffffu) << 16);            // [63:48] tensor_dim0 lo
  g1[2] = (int)(((tensor_d0 >> 16) & 0xffffu) | ((tensor_d1 & 0xffffu) << 16));
  g1[3] = (int)(((tensor_d1 >> 16) & 0xffffu) | ((tile_d0 & 0xffffu) << 16));
  g1[4] = (int)(tile_d1 & 0xffffu);                      // tile_dim1; tile_dim2=0
  g1[5] = (int)(unsigned)(row_stride_elems & 0xffffffffull);   // dim0_stride lo
  g1[6] = (int)(unsigned)((row_stride_elems >> 32) & 0xffffull); // dim0_stride hi; dim1_stride=0
  g1[7] = 0;

  v4i z4 = {0, 0, 0, 0};
  v8i z8 = {0, 0, 0, 0, 0, 0, 0, 0};
  // 6-arg variant (amdgpu-toolchain / clang-23 form).
  __builtin_amdgcn_tensor_load_to_lds(g0, g1, z4, z4, z8, 0);
}

// ---------------- f32 -> bf16 conversion ----------------
__global__ void f2bf_kernel(const float* __restrict__ s, unsigned short* __restrict__ d, long n){
  long i = (long)blockIdx.x * blockDim.x + threadIdx.x;
  long stride = (long)gridDim.x * blockDim.x;
  for (; i < n; i += stride) d[i] = f2bf(s[i]);
}

// ---------------- GEMM, flavor T: D = A * W^T (+bias)(+relu) ----------------
// Block 256 = 8 waves; block tile 64(m) x 256(n); wave tile 32x64.
__global__ __launch_bounds__(256) void gemm_t_kernel(
    const unsigned short* __restrict__ A, long lda,
    const unsigned short* __restrict__ W, long ldw,
    float* __restrict__ Df, unsigned short* __restrict__ Db, long ldd,
    const float* __restrict__ bias, int Kin, int relu)
{
  int tid = threadIdx.x, lid = tid & 31, w = tid >> 5;
  int lm = lid & 15, hi = lid >> 4;
  int wm = blockIdx.x * 64  + (w & 1)  * 32;
  int wn = blockIdx.y * 256 + (w >> 1) * 64;

  v8f acc[2][4];
#pragma unroll
  for (int i = 0; i < 2; ++i)
#pragma unroll
    for (int j = 0; j < 4; ++j) acc[i][j] = v8f_zero();

  for (int k0 = 0; k0 < Kin; k0 += 32){
    int kn = k0 + 32;
    if (kn < Kin){
      __builtin_prefetch(A + (long)(wm + lm) * lda + kn, 0, 3);
      __builtin_prefetch(W + (long)(wn + lm) * ldw + kn, 0, 3);
    }
    v16u a0 = load_a_frag(A, lda, wm,      k0, lid);
    v16u a1 = load_a_frag(A, lda, wm + 16, k0, lid);
#pragma unroll
    for (int j = 0; j < 4; ++j){
      v16u b = load_bT_frag(W, ldw, wn + j * 16, k0, lid);
      acc[0][j] = wmma_bf16(a0, b, acc[0][j]);
      acc[1][j] = wmma_bf16(a1, b, acc[1][j]);
    }
  }

#pragma unroll
  for (int i = 0; i < 2; ++i){
#pragma unroll
    for (int j = 0; j < 4; ++j){
      int col = wn + j * 16 + lm;
      float bv = bias ? bias[col] : 0.f;
#pragma unroll
      for (int r = 0; r < 8; ++r){
        int row = wm + i * 16 + r + 8 * hi;
        float v = acc[i][j][r] + bv;
        if (relu) v = fmaxf(v, 0.f);
        if (Df) Df[(long)row * ldd + col] = v;
        else    Db[(long)row * ldd + col] = f2bf(v);
      }
    }
  }
}

// ---------------- GEMM, flavor N: D = A * B (+bias per row), bf16 out ----------------
// B (K x 64) is staged 32-k-rows at a time into LDS by the Tensor Data Mover
// (tensor_load_to_lds, TENSORcnt), double-buffered: wave 0 issues one TDM op
// per k-step describing the whole strided tile; the D# pad fields give the
// 144B LDS pitch. Batched over z = b*Hinner + h.
// Block 256 = 8 waves; block tile 128(m) x 64(n); wave tile 32x32.
__global__ __launch_bounds__(256) void gemm_n_kernel(
    const unsigned short* __restrict__ A,  long lda, long sAb, long sAh,
    const unsigned short* __restrict__ Bm, long ldb, long sBb, long sBh,
    unsigned short* __restrict__ Dst,      long ldd, long sDb, long sDh,
    const float* __restrict__ biasRow, int Hinner, int Kin)
{
  constexpr int TP = 72;                      // padded LDS pitch (halves): 144B
  __shared__ __attribute__((aligned(16))) unsigned short Bs[2][32 * TP];

  int z = blockIdx.z, zb = z / Hinner, zh = z % Hinner;
  A   += (long)zb * sAb + (long)zh * sAh;
  Bm  += (long)zb * sBb + (long)zh * sBh;
  Dst += (long)zb * sDb + (long)zh * sDh;

  int tid = threadIdx.x, lid = tid & 31, w = tid >> 5;
  int lm = lid & 15, hi = lid >> 4;
  int wm  = blockIdx.x * 128 + (w >> 1) * 32;
  int wnl = (w & 1) * 32;                     // column offset local to the 64-wide tile
  int cb  = blockIdx.y * 64;                  // global column base

  unsigned lds0 = (unsigned)(unsigned long long)(void*)&Bs[0][0];
  unsigned bufbytes = 32u * TP * 2u;          // 4608 B per buffer
  unsigned long long bmb = (unsigned long long)Bm + (unsigned long long)cb * 2ull;

  v8f acc[2][2];
#pragma unroll
  for (int i = 0; i < 2; ++i)
#pragma unroll
    for (int j = 0; j < 2; ++j) acc[i][j] = v8f_zero();

  // prologue: DMA k-rows [0,32)
  if (w == 0){
    tdm_load_tile(lds0, bmb, 64u, 32u, 64u, 32u, (unsigned long long)ldb);
    __builtin_amdgcn_s_wait_tensorcnt(0);
  }
  __syncthreads();

  int buf = 0;
  for (int k0 = 0; k0 < Kin; k0 += 32){
    int kn = k0 + 32;
    if (kn < Kin && w == 0){                  // wave-uniform branch
      unsigned long long ga = bmb + (unsigned long long)((long)kn * ldb) * 2ull;
      tdm_load_tile(lds0 + (unsigned)(buf ^ 1) * bufbytes, ga,
                    64u, 32u, 64u, 32u, (unsigned long long)ldb);
    }

    v16u a0 = load_a_frag(A, lda, wm,      k0, lid);
    v16u a1 = load_a_frag(A, lda, wm + 16, k0, lid);

    v16u b0, b1;
#pragma unroll
    for (int j = 0; j < 16; ++j){
      int kk = 16 * hi + j;
      b0[j] = Bs[buf][kk * TP + wnl + lm];
      b1[j] = Bs[buf][kk * TP + wnl + 16 + lm];
    }
    acc[0][0] = wmma_bf16(a0, b0, acc[0][0]);
    acc[0][1] = wmma_bf16(a0, b1, acc[0][1]);
    acc[1][0] = wmma_bf16(a1, b0, acc[1][0]);
    acc[1][1] = wmma_bf16(a1, b1, acc[1][1]);

    if (kn < Kin){
      if (w == 0) __builtin_amdgcn_s_wait_tensorcnt(0);
      __syncthreads();
      buf ^= 1;
    }
  }

#pragma unroll
  for (int i = 0; i < 2; ++i){
#pragma unroll
    for (int j = 0; j < 2; ++j){
      int col = cb + wnl + j * 16 + lm;
#pragma unroll
      for (int r = 0; r < 8; ++r){
        int row = wm + i * 16 + r + 8 * hi;
        float v = acc[i][j][r] + (biasRow ? biasRow[row] : 0.f);
        Dst[(long)row * ldd + col] = f2bf(v);
      }
    }
  }
}

// ---------------- fused scores + softmax: P = softmax(scale * Q * KE^T) ----------------
// Block 128 = 4 waves; each block owns 16 rows x all 256 cols (row in registers).
__global__ __launch_bounds__(128) void scores_softmax_kernel(
    const unsigned short* __restrict__ Q,  long ldq,  long sQb,  long sQh,
    const unsigned short* __restrict__ KE, long ldke, long sKEb, long sKEh,
    unsigned short* __restrict__ P,        long ldp,  long sPb,  long sPh,
    int Hinner)
{
  int z = blockIdx.z, zb = z / Hinner, zh = z % Hinner;
  Q  += (long)zb * sQb  + (long)zh * sQh;
  KE += (long)zb * sKEb + (long)zh * sKEh;
  P  += (long)zb * sPb  + (long)zh * sPh;

  int tid = threadIdx.x, lid = tid & 31, w = tid >> 5;
  int lm = lid & 15, hi = lid >> 4;
  int m0 = blockIdx.x * 16;
  int n0 = w * 64;

  v8f acc[4];
#pragma unroll
  for (int t = 0; t < 4; ++t) acc[t] = v8f_zero();

  for (int k0 = 0; k0 < Dd; k0 += 32){
    v16u a = load_a_frag(Q, ldq, m0, k0, lid);
#pragma unroll
    for (int t = 0; t < 4; ++t){
      v16u b = load_bT_frag(KE, ldke, n0 + t * 16, k0, lid);
      acc[t] = wmma_bf16(a, b, acc[t]);
    }
  }

  __shared__ float smax[4][16];
  __shared__ float ssum[4][16];
  const float c = 0.125f * 1.4426950408889634f;   // (1/sqrt(D)) * log2(e)

  float tmp[8];
#pragma unroll
  for (int r = 0; r < 8; ++r){
    float m = fmaxf(fmaxf(acc[0][r], acc[1][r]), fmaxf(acc[2][r], acc[3][r]));
#pragma unroll
    for (int mk = 1; mk <= 8; mk <<= 1) m = fmaxf(m, __shfl_xor(m, mk, 32));
    tmp[r] = m;
  }
  if (lm == 0){
#pragma unroll
    for (int r = 0; r < 8; ++r) smax[w][r + 8 * hi] = tmp[r];
  }
  __syncthreads();

#pragma unroll
  for (int r = 0; r < 8; ++r){
    int rr = r + 8 * hi;
    float m = fmaxf(fmaxf(smax[0][rr], smax[1][rr]), fmaxf(smax[2][rr], smax[3][rr]));
    float s = 0.f;
#pragma unroll
    for (int t = 0; t < 4; ++t){
      float p = exp2f((acc[t][r] - m) * c);
      acc[t][r] = p;
      s += p;
    }
#pragma unroll
    for (int mk = 1; mk <= 8; mk <<= 1) s += __shfl_xor(s, mk, 32);
    tmp[r] = s;
  }
  if (lm == 0){
#pragma unroll
    for (int r = 0; r < 8; ++r) ssum[w][r + 8 * hi] = tmp[r];
  }
  __syncthreads();

#pragma unroll
  for (int r = 0; r < 8; ++r){
    int rr = r + 8 * hi;
    float inv = 1.0f / (ssum[0][rr] + ssum[1][rr] + ssum[2][rr] + ssum[3][rr]);
    int row = m0 + rr;
#pragma unroll
    for (int t = 0; t < 4; ++t)
      P[(long)row * ldp + n0 + t * 16 + lm] = f2bf(acc[t][r] * inv);
  }
}

// ---------------- residual + LayerNorm (C = 1024), emits f32 + bf16 ----------------
__global__ __launch_bounds__(256) void ln_residual_kernel(
    const float* __restrict__ res, const float* __restrict__ t,
    const float* __restrict__ g,   const float* __restrict__ bta,
    float* __restrict__ of, unsigned short* __restrict__ ob)
{
  const int Cn = Cc;
  long row = blockIdx.x;
  const float* r0 = res + row * Cn;
  const float* t0 = t   + row * Cn;
  int tid = threadIdx.x;

  float x[4]; float s = 0.f, s2 = 0.f;
#pragma unroll
  for (int i = 0; i < 4; ++i){
    int cix = tid + i * 256;
    float v = r0[cix] + t0[cix];
    x[i] = v; s += v; s2 += v * v;
  }
  __shared__ float rs[256], rq[256];
  rs[tid] = s; rq[tid] = s2;
  __syncthreads();
  for (int st = 128; st > 0; st >>= 1){
    if (tid < st){ rs[tid] += rs[tid + st]; rq[tid] += rq[tid + st]; }
    __syncthreads();
  }
  float mean = rs[0] * (1.0f / Cn);
  float var  = rq[0] * (1.0f / Cn) - mean * mean;
  float rinv = rsqrtf(var + 1e-5f);
#pragma unroll
  for (int i = 0; i < 4; ++i){
    int cix = tid + i * 256;
    float v = (x[i] - mean) * rinv * g[cix] + bta[cix];
    of[row * Cn + cix] = v;
    ob[row * Cn + cix] = f2bf(v);
  }
}

// ======================================================================
extern "C" void kernel_launch(void* const* d_in, const int* in_sizes, int n_in,
                              void* d_out, int out_size, void* d_ws, size_t ws_size,
                              hipStream_t stream)
{
  (void)in_sizes; (void)n_in; (void)out_size; (void)ws_size;

  const float* x_in = (const float*)d_in[0];
  const float* E_w  = (const float*)d_in[1];
  const float* E_b  = (const float*)d_in[2];
  const float* Wq   = (const float*)d_in[3];
  const float* Wk   = (const float*)d_in[4];
  const float* Wv   = (const float*)d_in[5];
  const float* wo_w = (const float*)d_in[6];
  const float* wo_b = (const float*)d_in[7];
  const float* ln1g = (const float*)d_in[8];
  const float* ln1b = (const float*)d_in[9];
  const float* ffw1 = (const float*)d_in[10];
  const float* ffb1 = (const float*)d_in[11];
  const float* ffw2 = (const float*)d_in[12];
  const float* ffb2 = (const float*)d_in[13];
  const float* ln2g = (const float*)d_in[14];
  const float* ln2b = (const float*)d_in[15];

  char* ws = (char*)d_ws;
  size_t off = 0;
  auto alloc = [&](size_t bytes) -> char* {
    size_t a = (off + 255) & ~(size_t)255;
    off = a + bytes;
    return ws + a;
  };

  const long BN  = (long)Bb * Nn;   // 16384 rows
  const long HD  = (long)Hh * Dd;   // 1024

  unsigned short* xb   = (unsigned short*)alloc((size_t)BN * Cc * 2);
  float*          xf   = (float*)         alloc((size_t)BN * Cc * 4);
  unsigned short* Ewb  = (unsigned short*)alloc((size_t)KD * Nn * 2);
  unsigned short* Wqb  = (unsigned short*)alloc((size_t)LL * HD * Cc * 2);
  unsigned short* Wkb  = (unsigned short*)alloc((size_t)LL * HD * Cc * 2);
  unsigned short* Wvb  = (unsigned short*)alloc((size_t)LL * HD * Cc * 2);
  unsigned short* wob  = (unsigned short*)alloc((size_t)LL * Cc * HD * 2);
  unsigned short* f1b  = (unsigned short*)alloc((size_t)LL * FFd * Cc * 2);
  unsigned short* f2b  = (unsigned short*)alloc((size_t)LL * Cc * FFd * 2);
  unsigned short* Qb   = (unsigned short*)alloc((size_t)BN * HD * 2);
  unsigned short* Kb   = (unsigned short*)alloc((size_t)BN * HD * 2);
  unsigned short* Vb   = (unsigned short*)alloc((size_t)BN * HD * 2);
  unsigned short* KEt  = (unsigned short*)alloc((size_t)Bb * Hh * KD * Dd * 2);
  unsigned short* VFt  = (unsigned short*)alloc((size_t)Bb * Hh * KD * Dd * 2);
  unsigned short* Pbuf = (unsigned short*)alloc((size_t)Bb * Hh * Nn * KD * 2); // reused as FFN hidden
  float*          tbuf = (float*)         alloc((size_t)BN * Cc * 4);
  unsigned short* catb = Qb;  // Q dead after scores -> reuse for concat(attn)

  auto conv = [&](const float* s, unsigned short* d, long n){
    f2bf_kernel<<<1024, 256, 0, stream>>>(s, d, n);
  };
  conv(x_in, xb,  BN * Cc);
  conv(E_w,  Ewb, (long)KD * Nn);
  conv(Wq,   Wqb, (long)LL * HD * Cc);
  conv(Wk,   Wkb, (long)LL * HD * Cc);
  conv(Wv,   Wvb, (long)LL * HD * Cc);
  conv(wo_w, wob, (long)LL * Cc * HD);
  conv(ffw1, f1b, (long)LL * FFd * Cc);
  conv(ffw2, f2b, (long)LL * Cc * FFd);

  auto gemmT = [&](const unsigned short* A, long lda,
                   const unsigned short* W, long ldw,
                   float* Df, unsigned short* Db, long ldd,
                   const float* bias, int M, int Ncols, int Kin, int relu){
    dim3 g((unsigned)(M / 64), (unsigned)(Ncols / 256), 1);
    gemm_t_kernel<<<g, 256, 0, stream>>>(A, lda, W, ldw, Df, Db, ldd, bias, Kin, relu);
  };

  const long sKEb = (long)Hh * KD * Dd, sKEh = (long)KD * Dd;
  const long sPb  = (long)Hh * Nn * KD, sPh  = (long)Nn * KD;

  for (int l = 0; l < LL; ++l){
    const unsigned short* wq  = Wqb + (long)l * HD * Cc;
    const unsigned short* wk  = Wkb + (long)l * HD * Cc;
    const unsigned short* wv  = Wvb + (long)l * HD * Cc;
    const unsigned short* wo  = wob + (long)l * Cc * HD;
    const unsigned short* fw1 = f1b + (long)l * FFd * Cc;
    const unsigned short* fw2 = f2b + (long)l * Cc * FFd;
    const float* xres = (l == 0) ? x_in : xf;

    // 1) Q/K/V projections: [BN,C] x [HD,C]^T -> bf16 [B,N,H*D]
    gemmT(xb, Cc, wq, Cc, nullptr, Qb, HD, nullptr, (int)BN, (int)HD, Cc, 0);
    gemmT(xb, Cc, wk, Cc, nullptr, Kb, HD, nullptr, (int)BN, (int)HD, Cc, 0);
    gemmT(xb, Cc, wv, Cc, nullptr, Vb, HD, nullptr, (int)BN, (int)HD, Cc, 0);

    // 2) KE^T / VF : [K_DIM,N] x K/V-head -> [B,H,K_DIM,D] (+E_b per row)
    {
      dim3 g((unsigned)(KD / 128), (unsigned)(Dd / 64), (unsigned)(Bb * Hh));
      gemm_n_kernel<<<g, 256, 0, stream>>>(Ewb, Nn, 0, 0,
                                           Kb, HD, (long)Nn * HD, (long)Dd,
                                           KEt, Dd, sKEb, sKEh,
                                           E_b, Hh, Nn);
      gemm_n_kernel<<<g, 256, 0, stream>>>(Ewb, Nn, 0, 0,
                                           Vb, HD, (long)Nn * HD, (long)Dd,
                                           VFt, Dd, sKEb, sKEh,
                                           E_b, Hh, Nn);
    }

    // 3) fused scores + softmax -> P bf16 [B,H,N,K_DIM]
    {
      dim3 g((unsigned)(Nn / 16), 1, (unsigned)(Bb * Hh));
      scores_softmax_kernel<<<g, 128, 0, stream>>>(
          Qb, HD, (long)Nn * HD, (long)Dd,
          KEt, Dd, sKEb, sKEh,
          Pbuf, KD, sPb, sPh, Hh);
    }

    // 4) attn = P x VF -> written directly in concat layout [B,N,H*D]
    {
      dim3 g((unsigned)(Nn / 128), (unsigned)(Dd / 64), (unsigned)(Bb * Hh));
      gemm_n_kernel<<<g, 256, 0, stream>>>(Pbuf, KD, sPb, sPh,
                                           VFt, Dd, sKEb, sKEh,
                                           catb, HD, (long)Nn * HD, (long)Dd,
                                           nullptr, Hh, KD);
    }

    // 5) output projection (+bias) -> f32
    gemmT(catb, HD, wo, HD, tbuf, nullptr, Cc, wo_b + (long)l * Cc, (int)BN, Cc, (int)HD, 0);

    // 6) residual + LN1
    ln_residual_kernel<<<(unsigned)BN, 256, 0, stream>>>(
        xres, tbuf, ln1g + (long)l * Cc, ln1b + (long)l * Cc, xf, xb);

    // 7) FF1 (+bias, ReLU) -> bf16 hidden (reuses P buffer)
    unsigned short* hbuf = Pbuf;
    gemmT(xb, Cc, fw1, Cc, nullptr, hbuf, FFd, ffb1 + (long)l * FFd, (int)BN, FFd, Cc, 1);

    // 8) FF2 (+bias) -> f32
    gemmT(hbuf, FFd, fw2, FFd, tbuf, nullptr, Cc, ffb2 + (long)l * Cc, (int)BN, Cc, FFd, 0);

    // 9) residual + LN2 (last layer writes d_out)
    float* oxf = (l == LL - 1) ? (float*)d_out : xf;
    ln_residual_kernel<<<(unsigned)BN, 256, 0, stream>>>(
        xf, tbuf, ln2g + (long)l * Cc, ln2b + (long)l * Cc, oxf, xb);
  }
}
